// Network_300647710734
// MI455X (gfx1250) — matrix operand order repaired and checked by reading, not compiled, not run on hardware
//
#include <hip/hip_runtime.h>
#include <hip/hip_bf16.h>

typedef __attribute__((ext_vector_type(16))) _Float16 v16h;
typedef __attribute__((ext_vector_type(8)))  _Float16 v8h;
typedef __attribute__((ext_vector_type(8)))  float    v8f;

// gcc-style 4xint vector to match the async-to-LDS builtin's parameter type
typedef int v4i_g __attribute__((vector_size(16)));
typedef __attribute__((address_space(1))) v4i_g* gp128;   // global int4*
typedef __attribute__((address_space(3))) v4i_g* lp128;   // LDS int4*

#define B_    32
#define T_    100
#define FIN   16384
#define HID   1024
#define OUT_  256
#define BT    (B_ * T_)

#define TAUM  5.0f
#define TAUS  3.0f
#define THR   1.0f

// Detect the gfx1250 async-to-LDS path; fall back to the proven sync path.
#if defined(__has_builtin)
#if __has_builtin(__builtin_amdgcn_global_load_async_to_lds_b128)
#define HAVE_ASYNC_LDS 1
#endif
#if __has_builtin(__builtin_amdgcn_s_wait_asynccnt)
#define HAVE_WAIT_ASYNC 1
#endif
#endif

// ---------------------------------------------------------------------------
// f32 -> f16 elementwise (weight conversion)
// ---------------------------------------------------------------------------
__global__ void cvt_f32_f16(const float* __restrict__ src,
                            _Float16* __restrict__ dst, int n) {
  int i = blockIdx.x * blockDim.x + threadIdx.x;
  if (i < n) dst[i] = (_Float16)src[i];
}

// ---------------------------------------------------------------------------
// spike [B, F, T] f32  ->  X1h [(t*B+b), F] f16   (LDS tile transpose)
// ---------------------------------------------------------------------------
__global__ __launch_bounds__(256)
void transpose_spikes(const float* __restrict__ spike,
                      _Float16* __restrict__ X1h) {
  __shared__ float tile[64][101];
  int f0 = blockIdx.x * 64;
  int b  = blockIdx.y;
  const float* src = spike + ((size_t)b * FIN + f0) * T_;
  for (int idx = threadIdx.x; idx < 64 * T_; idx += 256) {
    int fr = idx / T_, tt = idx % T_;      // consecutive tid -> consecutive t (coalesced)
    tile[fr][tt] = src[fr * T_ + tt];
  }
  __syncthreads();
  for (int idx = threadIdx.x; idx < 64 * T_; idx += 256) {
    int fr = idx & 63, tt = idx >> 6;      // consecutive tid -> consecutive f (coalesced)
    X1h[(size_t)(tt * B_ + b) * FIN + f0 + fr] = (_Float16)tile[fr][tt];
  }
}

// ---------------------------------------------------------------------------
// C[M,N] f32 = A[M,K] f16 x W[N,K]^T f16 via v_wmma_f32_16x16x32_f16
// block: 512 threads (16 waves), tile 128(M) x 256(N), K-step 32,
// LDS double-buffered, async global->LDS loads, 1 barrier per K-step.
// wave w: mw=(w&3)*32, nw=(w>>2)*64 -> 2x4 16x16 accumulators (8 wmma/step)
// ---------------------------------------------------------------------------
template <int KD>
__global__ __launch_bounds__(512)
void gemm_f16(const _Float16* __restrict__ A,
              const _Float16* __restrict__ W,
              float* __restrict__ C, int N) {
  __shared__ _Float16 As[2][128][40];   // 128 rows x 32 K (padded to 40)
  __shared__ _Float16 Bs[2][256][40];   // 256 cols(n) x 32 K (padded)

  const int tid  = threadIdx.x;
  const int lane = tid & 31;
  const int w    = tid >> 5;            // 0..15
  const int mw   = (w & 3) * 32;
  const int nw   = (w >> 2) * 64;
  const int lh   = lane & 15;
  const int hi   = lane >> 4;           // 0: lanes 0-15, 1: lanes 16-31

  const int rowBase = blockIdx.x * 128;
  const int colBase = blockIdx.y * 256;

  // cooperative load assignment: one b128 (8 halves) per chunk
  const int arow = tid >> 2, ak = (tid & 3) * 8;   // A: 128x32 -> 1 chunk/thread
  const int brow = tid >> 2, bk = (tid & 3) * 8;   // B: 256x32 -> 2 chunks/thread
  const _Float16* Ag = A + (size_t)(rowBase + arow) * KD + ak;
  const _Float16* Wg = W + (size_t)(colBase + brow) * KD + bk;
  const size_t WgStep = (size_t)128 * KD;

  v8f acc[2][4] = {};
  constexpr int NT = KD / 32;

  auto issue = [&](int k0, int buf) {
#ifdef HAVE_ASYNC_LDS
    __builtin_amdgcn_global_load_async_to_lds_b128(
        (gp128)(void*)(Ag + k0), (lp128)(void*)&As[buf][arow][ak], 0, 0);
    __builtin_amdgcn_global_load_async_to_lds_b128(
        (gp128)(void*)(Wg + k0), (lp128)(void*)&Bs[buf][brow][bk], 0, 0);
    __builtin_amdgcn_global_load_async_to_lds_b128(
        (gp128)(void*)(Wg + WgStep + k0), (lp128)(void*)&Bs[buf][brow + 128][bk], 0, 0);
#else
    v8h a  = *(const v8h*)(Ag + k0);
    v8h b0 = *(const v8h*)(Wg + k0);
    v8h b1 = *(const v8h*)(Wg + WgStep + k0);
    *(v8h*)&As[buf][arow][ak]       = a;
    *(v8h*)&Bs[buf][brow][bk]       = b0;
    *(v8h*)&Bs[buf][brow + 128][bk] = b1;
#endif
    __builtin_prefetch(Ag + k0 + 64, 0, 1);   // global_prefetch_b8, 2 tiles ahead
    __builtin_prefetch(Wg + k0 + 64, 0, 1);
  };

  issue(0, 0);

  for (int i = 0; i < NT; ++i) {
#ifdef HAVE_ASYNC_LDS
#ifdef HAVE_WAIT_ASYNC
    __builtin_amdgcn_s_wait_asynccnt(0);
#else
    asm volatile("s_wait_asynccnt 0" ::: "memory");
#endif
#endif
    __syncthreads();
    if (i + 1 < NT) issue((i + 1) * 32, (i + 1) & 1);

    const int buf = i & 1;
    union H16 { v16h v; v8h h[2]; };
    H16 af[2], bf[4];
#pragma unroll
    for (int sm = 0; sm < 2; ++sm) {
      // A frag (16x32): lane = row M=lh, halves [hi*8, hi*8+8) and +16
      const _Float16* ap = &As[buf][mw + sm * 16 + lh][hi * 8];
      af[sm].h[0] = *(const v8h*)ap;
      af[sm].h[1] = *(const v8h*)(ap + 16);
    }
#pragma unroll
    for (int sn = 0; sn < 4; ++sn) {
      // B frag (32x16): lane = col N=lh, halves [hi*8, hi*8+8) and +16
      const _Float16* bp = &Bs[buf][nw + sn * 16 + lh][hi * 8];
      bf[sn].h[0] = *(const v8h*)bp;
      bf[sn].h[1] = *(const v8h*)(bp + 16);
    }
#pragma unroll
    for (int sm = 0; sm < 2; ++sm)
#pragma unroll
      for (int sn = 0; sn < 4; ++sn)
        acc[sm][sn] = __builtin_amdgcn_wmma_f32_16x16x32_f16(
            false, af[sm].v, false, bf[sn].v, (short)0, acc[sm][sn], false, false);
  }

  // epilogue: VGPR r -> M = r + hi*8 ; N = lh
#pragma unroll
  for (int sm = 0; sm < 2; ++sm)
#pragma unroll
    for (int sn = 0; sn < 4; ++sn)
#pragma unroll
      for (int r = 0; r < 8; ++r) {
        int row = rowBase + mw + sm * 16 + hi * 8 + r;
        int col = colBase + nw + sn * 16 + lh;
        C[(size_t)row * N + col] = acc[sm][sn][r];
      }
}

// ---------------------------------------------------------------------------
// fused PSP + LIF scan over T (layer 1). One thread per (b, o).
// psp commutes with the linear layer, so it is applied post-GEMM here.
// ---------------------------------------------------------------------------
__global__ void psp_lif(const float* __restrict__ Y,
                        _Float16* __restrict__ Xo, int Nf) {
  int i = blockIdx.x * blockDim.x + threadIdx.x;
  if (i >= B_ * Nf) return;
  int b = i / Nf, o = i % Nf;
  const float ks = 1.0f / TAUS, km = 1.0f / TAUM;
  float synp = 0.f, mem = 0.f, syn = 0.f;
  for (int t = 0; t < T_; ++t) {
    float y = Y[(size_t)(t * B_ + b) * Nf + o];
    synp = synp - synp * ks + y;          // psp state
    float xin = synp * ks;                // psp output
    float mem_pre = mem - mem * km + xin; // LIF membrane
    float spike = mem_pre > THR ? 1.f : 0.f;
    syn = syn + (spike - syn) * ks;       // synaptic trace (output)
    mem = mem_pre * (1.f - spike);
    Xo[(size_t)(t * B_ + b) * Nf + o] = (_Float16)syn;
  }
}

// ---------------------------------------------------------------------------
// LIF scan over T (layer 2) writing final output [B, OUT, 1, 1, T] f32
// ---------------------------------------------------------------------------
__global__ void lif_out(const float* __restrict__ Y, float* __restrict__ out) {
  int i = blockIdx.x * blockDim.x + threadIdx.x;
  if (i >= B_ * OUT_) return;
  int b = i / OUT_, o = i % OUT_;
  const float ks = 1.0f / TAUS, km = 1.0f / TAUM;
  float mem = 0.f, syn = 0.f;
  for (int t = 0; t < T_; ++t) {
    float y = Y[(size_t)(t * B_ + b) * OUT_ + o];
    float mem_pre = mem - mem * km + y;
    float spike = mem_pre > THR ? 1.f : 0.f;
    syn = syn + (spike - syn) * ks;
    mem = mem_pre * (1.f - spike);
    out[((size_t)b * OUT_ + o) * T_ + t] = syn;
  }
}

// ---------------------------------------------------------------------------
extern "C" void kernel_launch(void* const* d_in, const int* in_sizes, int n_in,
                              void* d_out, int out_size, void* d_ws, size_t ws_size,
                              hipStream_t stream) {
  (void)in_sizes; (void)n_in; (void)out_size; (void)ws_size;
  const float* spike = (const float*)d_in[0];
  const float* W1    = (const float*)d_in[1];
  const float* W2    = (const float*)d_in[2];
  float* out = (float*)d_out;

  char* ws = (char*)d_ws;
  size_t off = 0;
  auto carve = [&](size_t bytes) -> void* {
    void* p = ws + off;
    off = (off + bytes + 255) & ~(size_t)255;
    return p;
  };
  _Float16* X1h = (_Float16*)carve((size_t)BT * FIN * 2);   // ~105 MB
  _Float16* W1h = (_Float16*)carve((size_t)HID * FIN * 2);  // ~34 MB
  _Float16* W2h = (_Float16*)carve((size_t)OUT_ * HID * 2);
  float*    Y1  = (float*)carve((size_t)BT * HID * 4);      // ~13 MB
  _Float16* X2h = (_Float16*)carve((size_t)BT * HID * 2);
  float*    Y2  = (float*)carve((size_t)BT * OUT_ * 4);

  cvt_f32_f16<<<(HID * FIN + 255) / 256, 256, 0, stream>>>(W1, W1h, HID * FIN);
  cvt_f32_f16<<<(OUT_ * HID + 255) / 256, 256, 0, stream>>>(W2, W2h, OUT_ * HID);
  transpose_spikes<<<dim3(FIN / 64, B_), 256, 0, stream>>>(spike, X1h);

  gemm_f16<FIN><<<dim3(BT / 128, HID / 256), 512, 0, stream>>>(X1h, W1h, Y1, HID);
  psp_lif<<<(B_ * HID + 255) / 256, 256, 0, stream>>>(Y1, X2h, HID);
  gemm_f16<HID><<<dim3(BT / 128, OUT_ / 256), 512, 0, stream>>>(X2h, W2h, Y2, OUT_);
  lif_out<<<(B_ * OUT_ + 255) / 256, 256, 0, stream>>>(Y2, out);
}
